// Net_58360015618509
// MI455X (gfx1250) — matrix operand order repaired and checked by reading, not compiled, not run on hardware
//
#include <hip/hip_runtime.h>
#include <hip/hip_bf16.h>
#include <math.h>

// Problem constants (match reference)
#define T_   512
#define B_   128
#define INP_ 256
#define HS_  512
#define OUT_ 256

// Persistent-kernel geometry: 32 WGs x 8 wave32 = 256 waves = 256 C-tiles (128x512 / 16x16)
#define NWG   32
#define NTHR  256
#define WAVES 8

// LDS row strides (bf16 elements). 1040B stride -> row r starts at bank (r*260)%64 = 4r:
// all 16 rows of a tile land on distinct bank offsets (no ds bank aliasing).
#define LDW   (HS_ + 8)    // 520 elements = 1040 B  (weights / act / h rows)
#define LDX   (INP_ + 8)   // 264 elements =  528 B  (x rows)

// LDS layout (elements): [per-wave Wih|Whh tiles][act block][h block][x block]
#define LDS_W_ELE    (WAVES * 2 * 16 * LDW)                 // 133120
#define LDS_ACT_OFF  (LDS_W_ELE)                            // +8320
#define LDS_H_OFF    (LDS_ACT_OFF + 16 * LDW)               // +8320
#define LDS_X_OFF    (LDS_H_OFF + 16 * LDW)                 // +4224
#define LDS_TOT_ELE  (LDS_X_OFF + 16 * LDX)                 // 153984 el = 307968 B < 320 KB

typedef __bf16 v16bf __attribute__((ext_vector_type(16)));
typedef __bf16 v8bf  __attribute__((ext_vector_type(8)));
typedef float  v8f   __attribute__((ext_vector_type(8)));

union ABf { v16bf v; v8bf h[2]; };

// ---------------------------------------------------------------- grid barrier
__device__ __forceinline__ void grid_sync(unsigned* c) {
  __syncthreads();
  if (threadIdx.x == 0) {
    __hip_atomic_fetch_add(c, 1u, __ATOMIC_RELEASE, __HIP_MEMORY_SCOPE_AGENT);
    while (__hip_atomic_load(c, __ATOMIC_ACQUIRE, __HIP_MEMORY_SCOPE_AGENT) < (unsigned)NWG)
      __builtin_amdgcn_s_sleep(1);
  }
  __syncthreads();
  __builtin_amdgcn_fence(__ATOMIC_ACQUIRE, "agent");
}

// A-operand (16x32 bf16), contiguous K (matches B): lane m<16 holds K[0..15]; m>=16 holds K[16..31]
__device__ __forceinline__ v16bf load_a_bf16(const __bf16* row, int k0, int half) {
  ABf a;
  a.h[0] = *(const v8bf*)(row + k0 + half * 16);
  a.h[1] = *(const v8bf*)(row + k0 + half * 16 + 8);
  return a.v;
}
// B-operand 32B fetched as two 16B halves (LDS rows are 1040B-strided: only 16B-aligned)
__device__ __forceinline__ v16bf load_b16x2(const __bf16* p) {
  ABf a;
  a.h[0] = *(const v8bf*)p;
  a.h[1] = *(const v8bf*)(p + 8);
  return a.v;
}

// Async L2->LDS 16B copy (ASYNCcnt path, no VGPR data movement)
__device__ __forceinline__ void async_copy16(unsigned lds_byte_off, const void* gptr) {
  asm volatile("global_load_async_to_lds_b128 %0, %1, off"
               :: "v"(lds_byte_off), "v"(gptr) : "memory");
}
__device__ __forceinline__ void async_wait0() {
  asm volatile("s_wait_asynccnt 0" ::: "memory");
}

// ---------------------------------------------------------------- setup: bf16 weights, zero state
__global__ void setup_kernel(const float* __restrict__ W1, const float* __restrict__ Wih,
                             const float* __restrict__ Whh, const float* __restrict__ Wo,
                             __bf16* w1b, __bf16* wihb, __bf16* whhb, __bf16* wob,
                             float* pot, __bf16* h0, unsigned* bars) {
  int i = blockIdx.x * blockDim.x + threadIdx.x;
  if (i < HS_ * INP_) w1b[i] = (__bf16)W1[i];
  if (i < HS_ * HS_)  { wihb[i] = (__bf16)Wih[i]; whhb[i] = (__bf16)Whh[i]; }
  if (i < OUT_ * HS_) wob[i] = (__bf16)Wo[i];
  if (i < B_ * HS_)   { pot[i] = 0.f; h0[i] = (__bf16)0.f; }
  if (i < 2 * T_)     bars[i] = 0u;
}

// ---------------------------------------------------------------- persistent recurrent kernel
__global__ void __launch_bounds__(NTHR, 1)
pglu_rnn_kernel(const float* __restrict__ data,
                const float* __restrict__ b1, const float* __restrict__ decay,
                const float* __restrict__ b_ih, const float* __restrict__ b_hh,
                const float* __restrict__ bo,
                const __bf16* __restrict__ w1b, const __bf16* __restrict__ wihb,
                const __bf16* __restrict__ whhb, const __bf16* __restrict__ wob,
                float* __restrict__ pot, __bf16* __restrict__ actb,
                __bf16* __restrict__ h0, __bf16* __restrict__ h1,
                unsigned* __restrict__ bars, float* __restrict__ out) {
  extern __shared__ __bf16 smem[];
  const int wid  = threadIdx.x >> 5;
  const int lane = threadIdx.x & 31;
  const int half = lane >> 4;
  const int lm   = lane & 15;
  const int tile = blockIdx.x * WAVES + wid;   // 0..255
  const int tb   = tile >> 5;                  // batch tile (0..7) -- identical for all waves of a WG
  const int tj   = tile & 31;                  // hidden-col tile (0..31)

  __bf16* my_ih   = smem + (size_t)wid * 2 * 16 * LDW;
  __bf16* my_hh   = my_ih + 16 * LDW;
  __bf16* lds_act = smem + LDS_ACT_OFF;
  __bf16* lds_h   = smem + LDS_H_OFF;
  __bf16* lds_x   = smem + LDS_X_OFF;

  // Stage this wave's 16x512 W_ih / W_hh slices into LDS once (reused 512x via ds_load_b128)
#pragma unroll
  for (int r = 0; r < 16; ++r) {
    const size_t grow = (size_t)(tj * 16 + r) * HS_;
    *(uint4*)(my_ih + r * LDW + lane * 16) = *(const uint4*)(wihb + grow + lane * 16);
    *(uint4*)(my_hh + r * LDW + lane * 16) = *(const uint4*)(whhb + grow + lane * 16);
  }
  __syncthreads();

  const int   colg = tj * 16 + lm;             // C-layout column
  const float b1v  = b1[colg];
  const float dkv  = decay[colg];
  const float bhv  = b_ih[colg] + b_hh[colg];

  const int crow0 = tb * 16 + 8 * half;        // C-layout base row (VGPR r -> row r + 8*half)
  float*  potp = pot  + (size_t)crow0 * HS_ + colg;
  __bf16* actp = actb + (size_t)crow0 * HS_ + colg;

  const __bf16* xrow_l   = lds_x   + lm * LDX;
  const __bf16* arow_act = lds_act + lm * LDW;
  const __bf16* arow_h   = lds_h   + lm * LDW;
  const __bf16* brow_ih  = my_ih + (size_t)lm * LDW + half * 16;
  const __bf16* brow_hh  = my_hh + (size_t)lm * LDW + half * 16;

  for (int t = 0; t < T_; ++t) {
    // -------- stage x_t row-block (16 x 256 f32 -> bf16) cooperatively, once per WG --------
    {
      const float* gx = data + ((size_t)t * B_ + tb * 16) * INP_;
#pragma unroll
      for (int k = 0; k < 2; ++k) {
        const int u   = k * NTHR + threadIdx.x;   // 512 units of 8 floats
        const int row = u >> 5;
        const int off = (u & 31) * 8;
        const float* p = gx + (size_t)row * INP_ + off;
        if (t + 1 < T_) __builtin_prefetch(p + (size_t)B_ * INP_, 0, 1);  // global_prefetch_b8
        ABf cv;
#pragma unroll
        for (int i = 0; i < 8; ++i) cv.h[0][i] = (__bf16)p[i];
        *(v8bf*)(lds_x + row * LDX + off) = cv.h[0];
      }
      __syncthreads();
    }

    // -------- phase A: u = x_t @ W1^T (software-pipelined A loads; W1 tile lives in VGPRs) --
    v8f u = {0.f, 0.f, 0.f, 0.f, 0.f, 0.f, 0.f, 0.f};
    {
      v16bf va = load_a_bf16(xrow_l, 0, half);
#pragma unroll
      for (int kk = 0; kk < INP_ / 32; ++kk) {
        v16bf va_n;
        if (kk + 1 < INP_ / 32) va_n = load_a_bf16(xrow_l, (kk + 1) * 32, half);
        v16bf vb = *(const v16bf*)(w1b + (size_t)colg * INP_ + kk * 32 + half * 16);
        u = __builtin_amdgcn_wmma_f32_16x16x32_bf16(false, va, false, vb, (short)0, u, false, false);
        if (kk + 1 < INP_ / 32) va = va_n;
      }
    }
#pragma unroll
    for (int r = 0; r < 8; ++r) {
      float pv = potp[(size_t)r * HS_] + u[r] + b1v;
      float av = pv > 0.f ? pv : 0.f;
      actp[(size_t)r * HS_] = (__bf16)av;
      potp[(size_t)r * HS_] = (pv - av) * dkv;
    }
    grid_sync(&bars[2 * t]);

    // -------- stage act & h row-blocks into LDS via async DMA (L2 -> LDS, 16KB each) ------
    const __bf16* hcur = (t & 1) ? h1 : h0;
    __bf16*       hnxt = (t & 1) ? h0 : h1;
    {
      const char* gact = (const char*)(actb + (size_t)(tb * 16) * HS_);
      const char* gh   = (const char*)(hcur + (size_t)(tb * 16) * HS_);
#pragma unroll
      for (int k = 0; k < 4; ++k) {
        const int u16 = k * NTHR + threadIdx.x;   // 1024 units of 16B (64 per 1KB row)
        const int row = u16 >> 6;
        const int c   = (u16 & 63) * 16;
        async_copy16((unsigned)(LDS_ACT_OFF * 2 + row * (LDW * 2) + c), gact + row * (HS_ * 2) + c);
        async_copy16((unsigned)(LDS_H_OFF   * 2 + row * (LDW * 2) + c), gh   + row * (HS_ * 2) + c);
      }
      async_wait0();
      __syncthreads();
    }

    // -------- phase B: h = tanh(act @ W_ih^T + h @ W_hh^T + b) ---------------------------
    // Single flattened 32-step K-loop, one-stage software pipeline on both LDS operands so
    // the backend can overlap ds_load latency with the WMMA pipe (counted s_wait_dscnt).
    v8f acc = {0.f, 0.f, 0.f, 0.f, 0.f, 0.f, 0.f, 0.f};
    {
      v16bf va = load_a_bf16(arow_act, 0, half);
      v16bf vb = load_b16x2(brow_ih);
#pragma unroll
      for (int kk = 0; kk < 32; ++kk) {
        v16bf va_n, vb_n;
        if (kk + 1 < 32) {
          const int kn = kk + 1;
          const __bf16* ar = (kn < 16) ? arow_act : arow_h;
          const __bf16* br = (kn < 16) ? brow_ih  : brow_hh;
          const int k0 = (kn & 15) * 32;
          va_n = load_a_bf16(ar, k0, half);
          vb_n = load_b16x2(br + k0);
        }
        acc = __builtin_amdgcn_wmma_f32_16x16x32_bf16(false, va, false, vb, (short)0, acc, false, false);
        if (kk + 1 < 32) { va = va_n; vb = vb_n; }
      }
    }
    __bf16* hout = hnxt + (size_t)crow0 * HS_ + colg;
#pragma unroll
    for (int r = 0; r < 8; ++r)
      hout[(size_t)r * HS_] = (__bf16)tanhf(acc[r] + bhv);
    grid_sync(&bars[2 * t + 1]);
  }

  // -------- final: out = h_last @ Wo^T + bo  (T even -> h_last lives in h0) --------
  if (tile < (B_ / 16) * (OUT_ / 16)) {        // 128 output tiles; wave-uniform branch
    const int ob  = tile >> 4, oc = tile & 15;
    const int col = oc * 16 + lm;
    const __bf16* hl_row = h0 + (size_t)(ob * 16 + lm) * HS_;
    v8f acc = {0.f, 0.f, 0.f, 0.f, 0.f, 0.f, 0.f, 0.f};
    {
      v16bf va = load_a_bf16(hl_row, 0, half);
      v16bf vb = *(const v16bf*)(wob + (size_t)col * HS_ + half * 16);
#pragma unroll
      for (int kk = 0; kk < HS_ / 32; ++kk) {
        v16bf va_n, vb_n;
        if (kk + 1 < HS_ / 32) {
          const int k0 = (kk + 1) * 32;
          va_n = load_a_bf16(hl_row, k0, half);
          vb_n = *(const v16bf*)(wob + (size_t)col * HS_ + k0 + half * 16);
        }
        acc = __builtin_amdgcn_wmma_f32_16x16x32_bf16(false, va, false, vb, (short)0, acc, false, false);
        if (kk + 1 < HS_ / 32) { va = va_n; vb = vb_n; }
      }
    }
    const float bov = bo[col];
    float* op = out + (size_t)(ob * 16 + 8 * half) * OUT_ + col;
#pragma unroll
    for (int r = 0; r < 8; ++r) op[(size_t)r * OUT_] = acc[r] + bov;
  }
}

// ---------------------------------------------------------------- host entry
extern "C" void kernel_launch(void* const* d_in, const int* in_sizes, int n_in,
                              void* d_out, int out_size, void* d_ws, size_t ws_size,
                              hipStream_t stream) {
  (void)in_sizes; (void)n_in; (void)out_size; (void)ws_size;
  const float* data = (const float*)d_in[0];
  const float* W1   = (const float*)d_in[1];
  const float* b1   = (const float*)d_in[2];
  const float* dec  = (const float*)d_in[3];
  const float* Wih  = (const float*)d_in[4];
  const float* Whh  = (const float*)d_in[5];
  const float* bih  = (const float*)d_in[6];
  const float* bhh  = (const float*)d_in[7];
  const float* Wo   = (const float*)d_in[8];
  const float* bo   = (const float*)d_in[9];

  char* ws = (char*)d_ws;
  __bf16*   w1b  = (__bf16*)(ws + 0);         //  256 KB (512x256 bf16)
  __bf16*   wihb = (__bf16*)(ws + 262144);    //  512 KB
  __bf16*   whhb = (__bf16*)(ws + 786432);    //  512 KB
  __bf16*   wob  = (__bf16*)(ws + 1310720);   //  256 KB
  __bf16*   actb = (__bf16*)(ws + 1572864);   //  128 KB (B x HS bf16)
  __bf16*   h0   = (__bf16*)(ws + 1703936);   //  128 KB
  __bf16*   h1   = (__bf16*)(ws + 1835008);   //  128 KB
  float*    pot  = (float*)(ws + 1966080);    //  256 KB (B x HS f32)
  unsigned* bars = (unsigned*)(ws + 2228224); //    4 KB (2*T counters)

  setup_kernel<<<1024, 256, 0, stream>>>(W1, Wih, Whh, Wo, w1b, wihb, whhb, wob, pot, h0, bars);

  const size_t lds_bytes = (size_t)LDS_TOT_ELE * sizeof(__bf16); // 307968 B < 320 KB/WGP
  pglu_rnn_kernel<<<NWG, NTHR, lds_bytes, stream>>>(
      data, b1, dec, bih, bhh, bo, w1b, wihb, whhb, wob,
      pot, actb, h0, h1, bars, (float*)d_out);
}
